// SelfAttention1D_69355131896262
// MI455X (gfx1250) — matrix operand order repaired and checked by reading, not compiled, not run on hardware
//
#include <hip/hip_runtime.h>

typedef __attribute__((ext_vector_type(16))) _Float16 v16h;
typedef __attribute__((ext_vector_type(8)))  float    v8f;
typedef __attribute__((ext_vector_type(4)))  int      v4i;
typedef __attribute__((ext_vector_type(4)))  float    v4f;

#define CH 64
#define LL 131072
#define NB 4

union FragH { v16h h; v4i q[2]; };

__device__ __forceinline__ v8f vzero8() {
    v8f v;
#pragma unroll
    for (int i = 0; i < 8; ++i) v[i] = 0.f;
    return v;
}

// B fragment (32x16): element S[K][N] stored as p[N*64 + K] (contiguous in K).
// Per-lane data = two contiguous 8-half runs -> two b128 LDS loads.
__device__ __forceinline__ v16h load_bT(const _Float16* p, int nbase, int kbase, int lane) {
    int off = (nbase + (lane & 15)) * 64 + kbase + (lane >> 4) * 8;
    FragH f;
    f.q[0] = *(const v4i*)(p + off);
    f.q[1] = *(const v4i*)(p + off + 16);
    return f.h;
}

// A fragment (16x32): element S[M][K] stored as p[M*64 + K].
// Two contiguous 8-half runs -> two b128 LDS loads.
__device__ __forceinline__ v16h load_aR(const _Float16* p, int mbase, int kbase, int lane) {
    int off = (mbase + (lane & 15)) * 64 + kbase + (lane >> 4) * 8;
    FragH f;
    f.q[0] = *(const v4i*)(p + off);
    f.q[1] = *(const v4i*)(p + off + 16);
    return f.h;
}

// A fragment from row-major f32 global source (row stride 64), converted to f16.
__device__ __forceinline__ v16h load_a_g32(const float* p, int mbase, int kbase, int lane) {
    int base = (mbase + (lane & 15)) * 64 + kbase + (lane >> 4) * 8;
    const v4f* p0 = (const v4f*)(p + base);
    const v4f* p1 = (const v4f*)(p + base + 16);
    v4f x0 = p0[0], x1 = p0[1], y0 = p1[0], y1 = p1[1];
    v16h a;
#pragma unroll
    for (int i = 0; i < 4; ++i) {
        a[i]      = (_Float16)x0[i];
        a[i + 4]  = (_Float16)x1[i];
        a[i + 8]  = (_Float16)y0[i];
        a[i + 12] = (_Float16)y1[i];
    }
    return a;
}

#define WMMA(A, Bf, Cf) __builtin_amdgcn_wmma_f32_16x16x32_f16(false, (A), false, (Bf), (short)0, (Cf), false, false)

// ---------------- kernel Z: zero accumulators ----------------
__global__ void kZ(float* p, int n) {
    int i = blockIdx.x * blockDim.x + threadIdx.x;
    if (i < n) p[i] = 0.f;
}

// ---------------- kernel A: q/k projections + dw + sumsq + Gram ----------------
__global__ __launch_bounds__(256) void kA(const float* __restrict__ x,
                                          const float* __restrict__ w_kv,
                                          const float* __restrict__ w_kv_dw,
                                          const float* __restrict__ w_q,
                                          const float* __restrict__ w_q_dw,
                                          float* __restrict__ G,
                                          float* __restrict__ sqq,
                                          float* __restrict__ sqk) {
    __shared__ alignas(16) _Float16 Xs[64 * 64];  // x tile, COLUMN-major: [col][ch]
    __shared__ alignas(16) float    Xh[2 * 64];   // halo cols, [h][ch]
    __shared__ alignas(16) float    Ys[128 * 66]; // Y = W x; rows 0..63 = k, 64..127 = q; col 0/65 halos
    __shared__ alignas(16) _Float16 Qs[64 * 64];  // q_raw [ch][col]
    __shared__ alignas(16) _Float16 Ks[64 * 64];  // k_raw [ch][col]

    int tid  = threadIdx.x;
    int lane = tid & 31;
    int wv   = tid >> 5;                 // 0..7 : M-tile id
    int b    = blockIdx.x >> 7;          // 128 blocks per batch
    int tile = blockIdx.x & 127;
    int xbase = b * CH * LL;

    const float* wrow = (wv < 4) ? w_kv : w_q;
    int mb = (wv < 4) ? wv * 16 : (wv - 4) * 16;
    v16h aw0 = load_a_g32(wrow, mb, 0, lane);
    v16h aw1 = load_a_g32(wrow, mb, 32, lane);

    int ch  = tid & 63;
    int grp = tid >> 6;
    float wk0 = w_kv_dw[ch * 3 + 0], wk1 = w_kv_dw[ch * 3 + 1], wk2 = w_kv_dw[ch * 3 + 2];
    float wq0 = w_q_dw[ch * 3 + 0],  wq1 = w_q_dw[ch * 3 + 1],  wq2 = w_q_dw[ch * 3 + 2];

    v8f g0 = vzero8(), g1 = vzero8();
    float ssq = 0.f, ssk = 0.f;

#pragma unroll 1
    for (int it = 0; it < 16; ++it) {
        int l0 = tile * 1024 + it * 64;
        // stage x tile: float4 global loads, transpose into column-major f16
#pragma unroll
        for (int i = 0; i < 4; ++i) {
            int idx = tid + i * 256;            // 0..1023
            int c4 = idx & 15, k = idx >> 4;    // 16 col-quads x 64 channels
            v4f xv = *(const v4f*)(x + xbase + k * LL + l0 + c4 * 4);
#pragma unroll
            for (int j = 0; j < 4; ++j) Xs[(c4 * 4 + j) * 64 + k] = (_Float16)xv[j];
        }
        if (tid < 128) {
            int h = tid >> 6, k = tid & 63;
            int col = h ? (l0 + 64) : (l0 - 1);
            Xh[h * 64 + k] = (col >= 0 && col < LL) ? x[xbase + k * LL + col] : 0.f;
        }
        __syncthreads();

        // Y = W * X  (M-tile = wv, 4 N-tiles, K=64)
#pragma unroll
        for (int nt = 0; nt < 4; ++nt) {
            v16h b0 = load_bT(Xs, nt * 16, 0, lane);
            v8f d = vzero8();
            d = WMMA(aw0, b0, d);
            v16h b1 = load_bT(Xs, nt * 16, 32, lane);
            d = WMMA(aw1, b1, d);
#pragma unroll
            for (int r = 0; r < 8; ++r) {
                int row = wv * 16 + r + (lane >> 4) * 8;
                Ys[row * 66 + 1 + nt * 16 + (lane & 15)] = d[r];
            }
        }
        // halo columns of Y via VALU dot products (one per thread)
        {
            int r = tid & 127, h = tid >> 7;
            const v4f* wr4 = (const v4f*)((r < 64) ? (w_kv + r * 64) : (w_q + (r - 64) * 64));
            const v4f* xh4 = (const v4f*)(Xh + h * 64);
            float acc = 0.f;
#pragma unroll
            for (int k = 0; k < 16; ++k) {
                v4f a = wr4[k], bv = xh4[k];
                acc += a[0] * bv[0] + a[1] * bv[1] + a[2] * bv[2] + a[3] * bv[3];
            }
            Ys[r * 66 + (h ? 65 : 0)] = acc;
        }
        __syncthreads();

        // depthwise conv -> Qs/Ks (f16) + sum-of-squares
#pragma unroll
        for (int cc = 0; cc < 16; ++cc) {
            int c = grp * 16 + cc;
            const float* yk = Ys + ch * 66 + c;
            const float* yq = Ys + (64 + ch) * 66 + c;
            float kr = wk0 * yk[0] + wk1 * yk[1] + wk2 * yk[2];
            float qr = wq0 * yq[0] + wq1 * yq[1] + wq2 * yq[2];
            Ks[ch * 64 + c] = (_Float16)kr;
            Qs[ch * 64 + c] = (_Float16)qr;
            ssk += kr * kr;
            ssq += qr * qr;
        }
        __syncthreads();

        // Gram: G += Q * K^T ; wave handles tiles 2wv, 2wv+1 of the 4x4 grid
#pragma unroll
        for (int tt = 0; tt < 2; ++tt) {
            int id = wv * 2 + tt;
            int mt = id >> 2, nt = id & 3;
            v16h aq0 = load_aR(Qs, mt * 16, 0, lane);
            v16h bk0 = load_bT(Ks, nt * 16, 0, lane);
            v16h aq1 = load_aR(Qs, mt * 16, 32, lane);
            v16h bk1 = load_bT(Ks, nt * 16, 32, lane);
            if (tt == 0) { g0 = WMMA(aq0, bk0, g0); g0 = WMMA(aq1, bk1, g0); }
            else         { g1 = WMMA(aq0, bk0, g1); g1 = WMMA(aq1, bk1, g1); }
        }
        __syncthreads();
    }

    atomicAdd(&sqk[b * 64 + ch], ssk);
    atomicAdd(&sqq[b * 64 + ch], ssq);
#pragma unroll
    for (int tt = 0; tt < 2; ++tt) {
        int id = wv * 2 + tt;
        int mt = id >> 2, nt = id & 3;
        v8f g = tt ? g1 : g0;
#pragma unroll
        for (int r = 0; r < 8; ++r) {
            int M = mt * 16 + r + (lane >> 4) * 8;
            int N = nt * 16 + (lane & 15);
            atomicAdd(&G[b * 4096 + M * 64 + N], g[r]);
        }
    }
}

// ---------------- kernel B: norms, softmax, M = W_proj * attn ----------------
__global__ __launch_bounds__(256) void kB(const float* __restrict__ G,
                                          const float* __restrict__ sqq,
                                          const float* __restrict__ sqk,
                                          const float* __restrict__ temp,
                                          const float* __restrict__ w_proj,
                                          float* __restrict__ Mmat) {
    __shared__ _Float16 attn[NB * 64 * 64];
    __shared__ float rks[NB * 64];
    int t = threadIdx.x;
    int b = t >> 6, c = t & 63;
    rks[t] = 1.f / fmaxf(sqrtf(sqk[t]), 1e-12f);
    __syncthreads();

    float rq = 1.f / fmaxf(sqrtf(sqq[t]), 1e-12f);
    float sc = temp[b] * rq;
    const float* grow = G + b * 4096 + c * 64;
    float mx = -1e30f;
    for (int d = 0; d < 64; ++d) mx = fmaxf(mx, grow[d] * sc * rks[b * 64 + d]);
    float s = 0.f;
    for (int d = 0; d < 64; ++d) {
        float e = __expf(grow[d] * sc * rks[b * 64 + d] - mx);
        s += e;
        attn[(b * 64 + c) * 64 + d] = (_Float16)e;
    }
    float inv = 1.f / s;
    for (int d = 0; d < 64; ++d)
        attn[(b * 64 + c) * 64 + d] = (_Float16)((float)attn[(b * 64 + c) * 64 + d] * inv);
    __syncthreads();

    int o = c;
    for (int d = 0; d < 64; ++d) {
        float acc = 0.f;
#pragma unroll 8
        for (int k = 0; k < 64; ++k) acc += w_proj[o * 64 + k] * (float)attn[(b * 64 + k) * 64 + d];
        Mmat[b * 4096 + o * 64 + d] = acc;
    }
}

// ---------------- kernel C: V = dw(W_v x), out = Mmat * V ----------------
__global__ __launch_bounds__(256) void kC(const float* __restrict__ x,
                                          const float* __restrict__ w_kv,
                                          const float* __restrict__ w_kv_dw,
                                          const float* __restrict__ Mmat,
                                          float* __restrict__ out) {
    __shared__ alignas(16) _Float16 Xs[64 * 64];  // column-major [col][ch]
    __shared__ alignas(16) float    Xh[2 * 64];
    __shared__ alignas(16) float    Ys[64 * 66];
    __shared__ alignas(16) _Float16 Vs[64 * 64];  // column-major [col][ch]

    int tid  = threadIdx.x;
    int lane = tid & 31;
    int wv   = tid >> 5;
    int b    = blockIdx.x >> 7;
    int tile = blockIdx.x & 127;
    int xbase = b * CH * LL;

    const float* wv_w = w_kv + 64 * 64;           // v rows of w_kv
    int mt  = wv & 3;
    int nt0 = (wv >> 2) * 2;
    v16h av0 = load_a_g32(wv_w, mt * 16, 0, lane);
    v16h av1 = load_a_g32(wv_w, mt * 16, 32, lane);
    v16h am0 = load_a_g32(Mmat + b * 4096, mt * 16, 0, lane);
    v16h am1 = load_a_g32(Mmat + b * 4096, mt * 16, 32, lane);

    int ch = tid & 63, grp = tid >> 6;
    float dw0 = w_kv_dw[(64 + ch) * 3 + 0], dw1 = w_kv_dw[(64 + ch) * 3 + 1], dw2 = w_kv_dw[(64 + ch) * 3 + 2];

#pragma unroll 1
    for (int it = 0; it < 16; ++it) {
        int l0 = tile * 1024 + it * 64;
#pragma unroll
        for (int i = 0; i < 4; ++i) {
            int idx = tid + i * 256;
            int c4 = idx & 15, k = idx >> 4;
            v4f xv = *(const v4f*)(x + xbase + k * LL + l0 + c4 * 4);
#pragma unroll
            for (int j = 0; j < 4; ++j) Xs[(c4 * 4 + j) * 64 + k] = (_Float16)xv[j];
        }
        if (tid < 128) {
            int h = tid >> 6, k = tid & 63;
            int col = h ? (l0 + 64) : (l0 - 1);
            Xh[h * 64 + k] = (col >= 0 && col < LL) ? x[xbase + k * LL + col] : 0.f;
        }
        __syncthreads();

        // Y = W_v * X
#pragma unroll
        for (int u = 0; u < 2; ++u) {
            int nt = nt0 + u;
            v16h b0 = load_bT(Xs, nt * 16, 0, lane);
            v8f d = vzero8();
            d = WMMA(av0, b0, d);
            v16h b1 = load_bT(Xs, nt * 16, 32, lane);
            d = WMMA(av1, b1, d);
#pragma unroll
            for (int r = 0; r < 8; ++r) {
                int row = mt * 16 + r + (lane >> 4) * 8;
                Ys[row * 66 + 1 + nt * 16 + (lane & 15)] = d[r];
            }
        }
        if (tid < 128) {
            int r = tid & 63, h = tid >> 6;
            const v4f* wr4 = (const v4f*)(wv_w + r * 64);
            const v4f* xh4 = (const v4f*)(Xh + h * 64);
            float acc = 0.f;
#pragma unroll
            for (int k = 0; k < 16; ++k) {
                v4f a = wr4[k], bv = xh4[k];
                acc += a[0] * bv[0] + a[1] * bv[1] + a[2] * bv[2] + a[3] * bv[3];
            }
            Ys[r * 66 + (h ? 65 : 0)] = acc;
        }
        __syncthreads();

        // depthwise -> Vs (f16, column-major)
#pragma unroll
        for (int cc = 0; cc < 16; ++cc) {
            int c = grp * 16 + cc;
            const float* yv = Ys + ch * 66 + c;
            Vs[c * 64 + ch] = (_Float16)(dw0 * yv[0] + dw1 * yv[1] + dw2 * yv[2]);
        }
        __syncthreads();

        // out = Mmat * V, coalesced f32 stores
#pragma unroll
        for (int u = 0; u < 2; ++u) {
            int nt = nt0 + u;
            v16h b0 = load_bT(Vs, nt * 16, 0, lane);
            v8f d = vzero8();
            d = WMMA(am0, b0, d);
            v16h b1 = load_bT(Vs, nt * 16, 32, lane);
            d = WMMA(am1, b1, d);
#pragma unroll
            for (int r = 0; r < 8; ++r) {
                int row = mt * 16 + r + (lane >> 4) * 8;
                int col = l0 + nt * 16 + (lane & 15);
                out[xbase + row * LL + col] = d[r];
            }
        }
        __syncthreads();
    }
}

extern "C" void kernel_launch(void* const* d_in, const int* in_sizes, int n_in,
                              void* d_out, int out_size, void* d_ws, size_t ws_size,
                              hipStream_t stream) {
    const float* x       = (const float*)d_in[0];
    const float* w_kv    = (const float*)d_in[1];
    const float* w_kv_dw = (const float*)d_in[2];
    const float* w_q     = (const float*)d_in[3];
    const float* w_q_dw  = (const float*)d_in[4];
    const float* w_proj  = (const float*)d_in[5];
    const float* temp    = (const float*)d_in[6];
    float* outp = (float*)d_out;

    float* W    = (float*)d_ws;
    float* G    = W;              // 4*64*64
    float* sqq  = W + 16384;      // 256
    float* sqk  = W + 16640;      // 256
    float* Mmat = W + 16896;      // 4*64*64

    kZ<<<(16896 + 255) / 256, 256, 0, stream>>>(W, 16896);
    kA<<<NB * 128, 256, 0, stream>>>(x, w_kv, w_kv_dw, w_q, w_q_dw, G, sqq, sqk);
    kB<<<1, 256, 0, stream>>>(G, sqq, sqk, temp, w_proj, Mmat);
    kC<<<NB * 128, 256, 0, stream>>>(x, w_kv, w_kv_dw, Mmat, outp);
}
